// RNN_test_26164940767384
// MI455X (gfx1250) — matrix operand order repaired and checked by reading, not compiled, not run on hardware
//
#include <hip/hip_runtime.h>
#include <hip/hip_bf16.h>
#include <math.h>

typedef __bf16 bf16_t;
typedef __attribute__((ext_vector_type(16))) __bf16 v16bf;
typedef __attribute__((ext_vector_type(8)))  __bf16 v8bf;
typedef __attribute__((ext_vector_type(8)))  float  v8f;

#define TSTEPS 500
#define DIN    700
#define NN1    50
#define NN2    100
#define BR     4
// padded GEMM geometry
#define K1W    768   // layer1 K: 752 -> 768
#define K1C    24    // 768/32 k-chunks
#define K2C    5     // layer2 K: 152 -> 160
#define KLC    7     // liquid K: 200 -> 224
#define T1TIL  13    // layer1 N: 200 -> 208 cols -> 13 tiles
#define T2TIL  25    // layer2 N: 400 -> 25 tiles
#define TLTIL  7     // liquid N: 100 -> 112 cols -> 7 tiles
#define SCW    128   // sC width (8 tiles per consume pass)

__device__ __forceinline__ float sigm(float v) { return 1.0f / (1.0f + __expf(-v)); }

// ---------------------------------------------------------------------------
// Pack (masked) weight matrix W[N][K] (row major) into wave32 WMMA B-fragment
// order for v_wmma_f32_16x16x32_bf16:
//   out[(((tt*kchunks)+kc)*32 + lane)*16 + j] = bf16( W[n][k] * mask[n][k] )
//   n = tt*16 + (lane&15),  k = kc*32 + ((lane>=16)?16:0) + j
// ---------------------------------------------------------------------------
__global__ void pack_weights(const float* __restrict__ W, const float* __restrict__ mask,
                             int N, int K, int kchunks, int total, bf16_t* __restrict__ out) {
  int idx = blockIdx.x * 256 + threadIdx.x;
  if (idx >= total) return;
  int j    = idx & 15;
  int lane = (idx >> 4) & 31;
  int rest = idx >> 9;
  int kc   = rest % kchunks;
  int tt   = rest / kchunks;
  int n = tt * 16 + (lane & 15);
  int k = kc * 32 + ((lane >> 4) << 4) + j;
  float v = 0.0f;
  if (n < N && k < K) {
    v = W[n * K + k];
    if (mask) v *= mask[n * K + k];
  }
  out[idx] = (bf16_t)v;
}

// ---------------------------------------------------------------------------
// One GEMM pass: waves round-robin over output N-tiles [tb, tb+tiles).
// A: LDS bf16, row stride K1W halves, A-fragment in contiguous k order (k = 16*(lane/16) + i).
// B: pre-packed global (L2-resident) fragments, 32B contiguous per lane.
// D: accumulated over kchunks in v8f, stored to sC[16][SCW] at local cols.
// Fragment loads are software-pipelined one k-chunk ahead so the loads for
// chunk k+1 are in flight while the WMMA for chunk k executes (this kernel is
// latency-bound: 8 WGs, serial 500-step recurrence, no occupancy to hide it).
// ---------------------------------------------------------------------------
__device__ __forceinline__
void gemm_pass(const bf16_t* __restrict__ Bp, const bf16_t* sA, float* sC,
               int tb, int tiles, int kchunks, int wave, int lane) {
  const int m    = lane & 15;
  const int koff = (lane >> 4) << 4;           // A frag: 0 or 16
  const int mr   = (lane >> 4) << 3;           // D frag row base: 0 or 8
  for (int tt = tb + wave; tt < tb + tiles; tt += 8) {
    v8f c = {};
    const bf16_t* bp = Bp + (size_t)tt * kchunks * 512 + lane * 16;
    const bf16_t* ap = sA + m * K1W + koff;
    // pipeline prologue: fragments for chunk 0
    v8bf  lo = *(const v8bf*)(ap);
    v8bf  hi = *(const v8bf*)(ap + 8);
    v16bf b  = *(const v16bf*)(bp);
#pragma unroll
    for (int kc = 0; kc < kchunks; ++kc) {
      v8bf  lo_n = lo;
      v8bf  hi_n = hi;
      v16bf b_n  = b;
      if (kc + 1 < kchunks) {                  // issue next chunk's loads early
        lo_n = *(const v8bf*)(ap + 32);
        hi_n = *(const v8bf*)(ap + 40);
        b_n  = *(const v16bf*)(bp + 512);
      }
      v16bf a;
#pragma unroll
      for (int i = 0; i < 8; ++i) { a[i] = lo[i]; a[i + 8] = hi[i]; }
      c = __builtin_amdgcn_wmma_f32_16x16x32_bf16(false, a, false, b, (short)0, c,
                                                  false, false);
      lo = lo_n; hi = hi_n; b = b_n;
      ap += 32;
      bp += 512;
    }
    const int col = (tt - tb) * 16 + m;
#pragma unroll
    for (int r = 0; r < 8; ++r) sC[(mr + r) * SCW + col] = c[r];
  }
}

// ---------------------------------------------------------------------------
// Persistent kernel: 8 workgroups x 256 threads; WG w owns batch rows
// [16w, 16w+16) and iterates the full 500-step recurrence.
// ---------------------------------------------------------------------------
__global__ __launch_bounds__(256, 1)
void snn_persistent(const float* __restrict__ x,
                    const float* __restrict__ b1g,  const float* __restrict__ tau_m1,
                    const float* __restrict__ tau_n1,
                    const float* __restrict__ b2g,  const float* __restrict__ tau_m2,
                    const float* __restrict__ tau_n2,
                    const float* __restrict__ bxg,  const float* __restrict__ btmg,
                    const float* __restrict__ btag,
                    const float* __restrict__ wrg,  const float* __restrict__ brg,
                    const float* __restrict__ tau_mr,
                    const float* __restrict__ mem1_0, const float* __restrict__ mem2_0,
                    const float* __restrict__ mem3_0, const float* __restrict__ memr_0,
                    const bf16_t* __restrict__ W1p, const bf16_t* __restrict__ W2p,
                    const bf16_t* __restrict__ WXp, const bf16_t* __restrict__ WTMp,
                    const bf16_t* __restrict__ WTAp,
                    float* __restrict__ out) {
  __shared__ __align__(16) bf16_t sK[16 * K1W];     // GEMM A staging (bf16)
  __shared__ __align__(16) float  sC[16 * SCW];     // GEMM D staging (f32)
  __shared__ bf16_t sSpk1[16 * NN1];
  __shared__ bf16_t sSpk2[16 * NN2];
  __shared__ bf16_t sSpk3[16 * NN2];
  __shared__ float  sMem3[16 * NN2];
  __shared__ float  sB3[16 * NN2];
  __shared__ float  sDx[16 * NN2];
  __shared__ float  sAlpha1[NN1], sBeta1[NN1 * BR];
  __shared__ float  sAlpha2[NN2], sBeta2[NN2 * BR];
  __shared__ float  sBx[NN2], sBtm[NN2], sBta[NN2];
  __shared__ float  sAccT[32];

  const int tid  = threadIdx.x;
  const int wave = tid >> 5;
  const int lane = tid & 31;
  const int wg   = blockIdx.x;           // 0..7, batch base = 16*wg

  // ---- per-thread register state ----
  float d1r[4][BR], mem1r[4], spk1r[4];          // 800 (b,n) pairs, 4 slots
  float d2r[7][BR], mem2r[7], spk2r[7];          // 1600 (b,n) pairs, 7 slots
  float dxr[7], tmr[7];
  float memrr = 0.0f, accr = 0.0f, alr = 0.0f, brr = 0.0f;

  // ---- init constants + state ----
  for (int i = tid; i < NN1; i += 256)        sAlpha1[i] = sigm(tau_m1[i]);
  for (int i = tid; i < NN1 * BR; i += 256)   sBeta1[i]  = sigm(tau_n1[i]);
  for (int i = tid; i < NN2; i += 256)        sAlpha2[i] = sigm(tau_m2[i]);
  for (int i = tid; i < NN2 * BR; i += 256)   sBeta2[i]  = sigm(tau_n2[i]);
  for (int i = tid; i < NN2; i += 256) { sBx[i] = bxg[i]; sBtm[i] = btmg[i]; sBta[i] = btag[i]; }
  for (int i = tid; i < 16 * NN2; i += 256) {
    sMem3[i] = mem3_0[wg * 16 * NN2 + i];
    sB3[i]   = 0.1f;                  // B_J0
    sSpk3[i] = (bf16_t)0.0f;
    sSpk2[i] = (bf16_t)0.0f;
  }
  for (int i = tid; i < 16 * NN1; i += 256) sSpk1[i] = (bf16_t)0.0f;
#pragma unroll
  for (int s = 0; s < 4; ++s) {
    int p = tid + s * 256;
    mem1r[s] = 0.0f; spk1r[s] = 0.0f;
#pragma unroll
    for (int j = 0; j < BR; ++j) d1r[s][j] = 0.0f;
    if (p < 16 * NN1) mem1r[s] = mem1_0[wg * 16 * NN1 + p];
  }
#pragma unroll
  for (int s = 0; s < 7; ++s) {
    int p = tid + s * 256;
    mem2r[s] = 0.0f; spk2r[s] = 0.0f; dxr[s] = 0.0f; tmr[s] = 0.0f;
#pragma unroll
    for (int j = 0; j < BR; ++j) d2r[s][j] = 0.0f;
    if (p < 16 * NN2) mem2r[s] = mem2_0[wg * 16 * NN2 + p];
  }
  if (tid < 32) {
    memrr = memr_0[wg * 32 + tid];
    alr   = sigm(tau_mr[tid & 1]);
    brr   = brg[tid & 1];
  }
  __syncthreads();

  // ================= time loop =================
  for (int t = 0; t < TSTEPS; ++t) {
    // ---- build k1 = [x_t | spk1 | pad] as bf16 in sK ----
    for (int idx = tid; idx < 16 * K1W; idx += 256) {
      int b = idx / K1W, c = idx % K1W;
      float v = 0.0f;
      if (c < DIN)      v = x[((size_t)(wg * 16 + b) * TSTEPS + t) * DIN + c];
      else if (c < DIN + NN1) v = (float)sSpk1[b * NN1 + (c - DIN)];
      sK[idx] = (bf16_t)v;
    }
    // prefetch next step's x slab into cache
    if (t + 1 < TSTEPS) {
      for (int l = tid; l < 16 * 22; l += 256) {
        int b = l / 22, cc = (l % 22) * 32;
        if (cc < DIN)
          __builtin_prefetch(&x[((size_t)(wg * 16 + b) * TSTEPS + (t + 1)) * DIN + cc], 0, 1);
      }
    }
    __syncthreads();

    // ---- layer 1: GEMM in 2 passes of <=8 tiles, consume after each ----
    for (int pass = 0; pass < 2; ++pass) {
      int tb = pass * 8;
      int tiles = (pass == 0) ? 8 : (T1TIL - 8);
      gemm_pass(W1p, sK, sC, tb, tiles, K1C, wave, lane);
      __syncthreads();
      int n0 = tb * 4, n1 = n0 + tiles * 4; if (n1 > NN1) n1 = NN1;
#pragma unroll
      for (int s = 0; s < 4; ++s) {
        int p = tid + s * 256;
        if (p < 16 * NN1) {
          int b = p / NN1, n = p % NN1;
          if (n >= n0 && n < n1) {
            float dsum = 0.0f;
#pragma unroll
            for (int j = 0; j < BR; ++j) {
              float be  = sBeta1[n * BR + j];
              float cur = sC[b * SCW + (n - n0) * BR + j] + b1g[n * BR + j];
              d1r[s][j] = be * d1r[s][j] + (1.0f - be) * cur;
              dsum += d1r[s][j];
            }
            float a = sAlpha1[n];
            mem1r[s] = mem1r[s] * a + (1.0f - a) * dsum - 1.0f * spk1r[s];
            spk1r[s] = (mem1r[s] - 1.0f) > 0.0f ? 1.0f : 0.0f;
            sSpk1[b * NN1 + n] = (bf16_t)spk1r[s];
          }
        }
      }
      __syncthreads();
    }

    // ---- build k2 = [spk1 | spk2 | pad] ----
    for (int idx = tid; idx < 16 * 160; idx += 256) {
      int b = idx / 160, c = idx % 160;
      bf16_t v = (bf16_t)0.0f;
      if (c < NN1)            v = sSpk1[b * NN1 + c];
      else if (c < NN1 + NN2) v = sSpk2[b * NN2 + (c - NN1)];
      sK[b * K1W + c] = v;
    }
    __syncthreads();

    // ---- layer 2: GEMM in 4 passes ----
    for (int pass = 0; pass < 4; ++pass) {
      int tb = pass * 8;
      int tiles = (T2TIL - tb < 8) ? (T2TIL - tb) : 8;
      gemm_pass(W2p, sK, sC, tb, tiles, K2C, wave, lane);
      __syncthreads();
      int n0 = tb * 4, n1 = n0 + tiles * 4; if (n1 > NN2) n1 = NN2;
#pragma unroll
      for (int s = 0; s < 7; ++s) {
        int p = tid + s * 256;
        if (p < 16 * NN2) {
          int b = p / NN2, n = p % NN2;
          if (n >= n0 && n < n1) {
            float dsum = 0.0f;
#pragma unroll
            for (int j = 0; j < BR; ++j) {
              float be  = sBeta2[n * BR + j];
              float cur = sC[b * SCW + (n - n0) * BR + j] + b2g[n * BR + j];
              d2r[s][j] = be * d2r[s][j] + (1.0f - be) * cur;
              dsum += d2r[s][j];
            }
            float a = sAlpha2[n];
            mem2r[s] = mem2r[s] * a + (1.0f - a) * dsum - 1.0f * spk2r[s];
            spk2r[s] = (mem2r[s] - 1.0f) > 0.0f ? 1.0f : 0.0f;
            sSpk2[b * NN2 + n] = (bf16_t)spk2r[s];
          }
        }
      }
      __syncthreads();
    }

    // ---- liquid cell: dx = [spk2|spk3] @ wx^T + bx ----
    for (int idx = tid; idx < 16 * 224; idx += 256) {
      int b = idx / 224, c = idx % 224;
      bf16_t v = (bf16_t)0.0f;
      if (c < NN2)            v = sSpk2[b * NN2 + c];
      else if (c < 2 * NN2)   v = sSpk3[b * NN2 + (c - NN2)];
      sK[b * K1W + c] = v;
    }
    __syncthreads();
    gemm_pass(WXp, sK, sC, 0, TLTIL, KLC, wave, lane);
    __syncthreads();
#pragma unroll
    for (int s = 0; s < 7; ++s) {
      int p = tid + s * 256;
      if (p < 16 * NN2) {
        int b = p / NN2, n = p % NN2;
        dxr[s] = sC[b * SCW + n] + sBx[n];
        sDx[b * NN2 + n] = dxr[s];
      }
    }
    __syncthreads();

    // ---- tm = sigmoid([dx|mem3] @ wtm^T + btm) ----
    for (int idx = tid; idx < 16 * 224; idx += 256) {
      int b = idx / 224, c = idx % 224;
      float v = 0.0f;
      if (c < NN2)          v = sDx[b * NN2 + c];
      else if (c < 2 * NN2) v = sMem3[b * NN2 + (c - NN2)];
      sK[b * K1W + c] = (bf16_t)v;
    }
    __syncthreads();
    gemm_pass(WTMp, sK, sC, 0, TLTIL, KLC, wave, lane);
    __syncthreads();
#pragma unroll
    for (int s = 0; s < 7; ++s) {
      int p = tid + s * 256;
      if (p < 16 * NN2) {
        int b = p / NN2, n = p % NN2;
        tmr[s] = sigm(sC[b * SCW + n] + sBtm[n]);
      }
    }
    __syncthreads();

    // ---- ta = sigmoid([dx|b3] @ wta^T + bta), then liquid state update ----
    for (int idx = tid; idx < 16 * 224; idx += 256) {
      int b = idx / 224, c = idx % 224;
      float v = 0.0f;
      if (c < NN2)          v = sDx[b * NN2 + c];
      else if (c < 2 * NN2) v = sB3[b * NN2 + (c - NN2)];
      sK[b * K1W + c] = (bf16_t)v;
    }
    __syncthreads();
    gemm_pass(WTAp, sK, sC, 0, TLTIL, KLC, wave, lane);
    __syncthreads();
#pragma unroll
    for (int s = 0; s < 7; ++s) {
      int p = tid + s * 256;
      if (p < 16 * NN2) {
        int b = p / NN2, n = p % NN2;
        int i = b * NN2 + n;
        float ta  = sigm(sC[b * SCW + n] + sBta[n]);
        float b3o = sB3[i];
        float s3o = (float)sSpk3[i];
        float m3o = sMem3[i];
        float b3n = ta * b3o + (1.0f - ta) * s3o;
        float Bth = 0.1f + 1.8f * b3n;
        float m3n = m3o * tmr[s] + (1.0f - tmr[s]) * dxr[s] - Bth * s3o;
        float s3n = (m3n - Bth) > 0.0f ? 1.0f : 0.0f;
        sMem3[i] = m3n; sB3[i] = b3n; sSpk3[i] = (bf16_t)s3n;
      }
    }
    __syncthreads();

    // ---- leaky readout (threads 0..31: (b,o) pairs) ----
    if (tid < 32) {
      int b = tid >> 1, o = tid & 1;
      float dot = 0.0f;
      for (int n = 0; n < NN2; ++n)
        dot += (float)sSpk3[b * NN2 + n] * wrg[o * NN2 + n];
      memrr = memrr * alr + (1.0f - alr) * (dot + brr);
      if (t > 0) accr += memrr;
    }
    __syncthreads();
  }

  // ---- softmax(acc / T) ----
  if (tid < 32) sAccT[tid] = accr;
  __syncthreads();
  if (tid < 16) {
    float z0 = sAccT[tid * 2 + 0] / (float)TSTEPS;
    float z1 = sAccT[tid * 2 + 1] / (float)TSTEPS;
    float mx = fmaxf(z0, z1);
    float e0 = __expf(z0 - mx), e1 = __expf(z1 - mx);
    float inv = 1.0f / (e0 + e1);
    out[(wg * 16 + tid) * 2 + 0] = e0 * inv;
    out[(wg * 16 + tid) * 2 + 1] = e1 * inv;
  }
}

// ---------------------------------------------------------------------------
extern "C" void kernel_launch(void* const* d_in, const int* in_sizes, int n_in,
                              void* d_out, int out_size, void* d_ws, size_t ws_size,
                              hipStream_t stream) {
  const float* x      = (const float*)d_in[0];
  const float* mask1  = (const float*)d_in[1];
  const float* w1     = (const float*)d_in[2];
  const float* b1     = (const float*)d_in[3];
  const float* tau_m1 = (const float*)d_in[4];
  const float* tau_n1 = (const float*)d_in[5];
  const float* mask2  = (const float*)d_in[6];
  const float* w2     = (const float*)d_in[7];
  const float* b2     = (const float*)d_in[8];
  const float* tau_m2 = (const float*)d_in[9];
  const float* tau_n2 = (const float*)d_in[10];
  const float* wx     = (const float*)d_in[11];
  const float* bx     = (const float*)d_in[12];
  const float* wtm    = (const float*)d_in[13];
  const float* btm    = (const float*)d_in[14];
  const float* wta    = (const float*)d_in[15];
  const float* bta    = (const float*)d_in[16];
  const float* wr     = (const float*)d_in[17];
  const float* br     = (const float*)d_in[18];
  const float* tau_mr = (const float*)d_in[19];
  const float* mem1_0 = (const float*)d_in[20];
  const float* mem2_0 = (const float*)d_in[21];
  const float* mem3_0 = (const float*)d_in[22];
  const float* memr_0 = (const float*)d_in[23];

  // workspace layout (bf16 halves)
  bf16_t* ws  = (bf16_t*)d_ws;
  const int szW1 = T1TIL * K1C * 512;   // 159744
  const int szW2 = T2TIL * K2C * 512;   //  64000
  const int szWL = TLTIL * KLC * 512;   //  25088
  bf16_t* W1p  = ws;
  bf16_t* W2p  = W1p + szW1;
  bf16_t* WXp  = W2p + szW2;
  bf16_t* WTMp = WXp + szWL;
  bf16_t* WTAp = WTMp + szWL;

  pack_weights<<<(szW1 + 255) / 256, 256, 0, stream>>>(w1, mask1, 200, 752, K1C, szW1, W1p);
  pack_weights<<<(szW2 + 255) / 256, 256, 0, stream>>>(w2, mask2, 400, 152, K2C, szW2, W2p);
  pack_weights<<<(szWL + 255) / 256, 256, 0, stream>>>(wx,  nullptr, 100, 200, KLC, szWL, WXp);
  pack_weights<<<(szWL + 255) / 256, 256, 0, stream>>>(wtm, nullptr, 100, 200, KLC, szWL, WTMp);
  pack_weights<<<(szWL + 255) / 256, 256, 0, stream>>>(wta, nullptr, 100, 200, KLC, szWL, WTAp);

  snn_persistent<<<8, 256, 0, stream>>>(
      x, b1, tau_m1, tau_n1, b2, tau_m2, tau_n2,
      bx, btm, bta, wr, br, tau_mr,
      mem1_0, mem2_0, mem3_0, memr_0,
      W1p, W2p, WXp, WTMp, WTAp,
      (float*)d_out);
}
